// AttnDecoder_24833500905707
// MI455X (gfx1250) — matrix operand order repaired and checked by reading, not compiled, not run on hardware
//
#include <hip/hip_runtime.h>
#include <math.h>
#include <stdint.h>

// Problem constants (reference: B=8192, T=10, H=CH=512)
#define BSZ 8192
#define TT  10
#define HH  512
#define G4  2048   // 4*H
#define NK  (HH / 32)

typedef __bf16 bf16;
typedef __attribute__((ext_vector_type(16))) __bf16 v16bf;
typedef __attribute__((ext_vector_type(8)))  __bf16 bf16x8;
typedef __attribute__((ext_vector_type(8)))  float  v8f;
typedef __attribute__((ext_vector_type(4)))  float  f32x4;
typedef int i32x4 __attribute__((vector_size(16)));   // matches builtin param type

union V16 { v16bf v; bf16x8 h[2]; };

template <bool B_> struct BoolC { static constexpr bool value = B_; };

__device__ __forceinline__ float sigf(float x) { return 1.0f / (1.0f + __expf(-x)); }

// ---- CDNA5 async global->LDS path (ASYNCcnt), with sync fallback ----------
// Builtin signature (per hipcc diagnostic): (int4* global_src, int4* lds_dst,
// imm offset, imm cpol) with generic->__device__/__shared__ implicit casts.
#if __has_builtin(__builtin_amdgcn_global_load_async_to_lds_b128)
#define HAVE_ASYNC_LDS 1
__device__ __forceinline__ void async_copy16(const void* g, void* l) {
    __builtin_amdgcn_global_load_async_to_lds_b128((i32x4*)g, (i32x4*)l, 0, 0);
}
__device__ __forceinline__ void wait_async0() {
#if __has_builtin(__builtin_amdgcn_s_wait_asynccnt)
    __builtin_amdgcn_s_wait_asynccnt(0);
#else
    asm volatile("s_wait_asynccnt 0x0" ::: "memory");
#endif
}
#else
#define HAVE_ASYNC_LDS 0
__device__ __forceinline__ void wait_async0() {}
#endif

// ---------------------------------------------------------------------------
// Prep 1: W_hh (2048x512 f32, row-major [n][k]) -> bf16 (same layout).
// ---------------------------------------------------------------------------
__global__ void k_prep_w(const float* __restrict__ whh, bf16* __restrict__ out) {
    int idx = (blockIdx.x * 256 + threadIdx.x) * 4;   // 2048*512 exact
    f32x4 v = *(const f32x4*)(whh + idx);
    out[idx + 0] = (bf16)v.x;
    out[idx + 1] = (bf16)v.y;
    out[idx + 2] = (bf16)v.z;
    out[idx + 3] = (bf16)v.w;
}

// ---------------------------------------------------------------------------
// Prep 2: bias = b_ih+b_hh; fc1/fc2 collapse: v[j] = sum_n fc2_w[n]*fc1_w[n,j]
// ---------------------------------------------------------------------------
__global__ void k_prep_misc(const float* __restrict__ b_ih, const float* __restrict__ b_hh,
                            const float* __restrict__ fc1_w, const float* __restrict__ fc1_b,
                            const float* __restrict__ fc2_w, const float* __restrict__ fc2_b,
                            float* __restrict__ bias, float* __restrict__ vvec,
                            float* __restrict__ c0) {
    int g = blockIdx.x * blockDim.x + threadIdx.x;    // 2 x 1024
    if (g < G4) bias[g] = b_ih[g] + b_hh[g];
    if (g < 2 * HH) {
        float acc = 0.f;
        for (int n = 0; n < HH; ++n) acc += fc2_w[n] * fc1_w[n * (2 * HH) + g];
        vvec[g] = acc;
    }
    if (g == 0) {
        float acc = fc2_b[0];
        for (int n = 0; n < HH; ++n) acc += fc2_w[n] * fc1_b[n];
        *c0 = acc;
    }
}

// ---------------------------------------------------------------------------
// Main fused kernel: [64 batch x 64 hidden] tile; i/f/g gate GEMMs via
// v_wmma_f32_16x16x32_bf16 each step (o only at t=T-1), c-scan in registers.
// Double-buffered LDS, one barrier per k-step, W staged via async DMA,
// h loaded one step ahead into registers (f32->bf16 cvt on store).
// ---------------------------------------------------------------------------
#define BM  64
#define BN  64
#define LDH 40   // padded bf16 stride (16B-aligned rows)

__global__ void k_main(const float* __restrict__ h, const float* __restrict__ y_seq,
                       const float* __restrict__ W_ih, const bf16* __restrict__ whh,
                       const float* __restrict__ bias, float* __restrict__ d_state) {
    __shared__ __align__(16) bf16 lds_h[2][BM * LDH];
    __shared__ __align__(16) bf16 lds_w[2][4 * BN * LDH];
    __shared__ float lds_y[TT * BM];
    __shared__ float lds_bias[4 * BN];
    __shared__ float lds_wih[4 * BN];

    const int tid  = threadIdx.x;
    const int lane = tid & 31;
    const int wid  = tid >> 5;
    const int wm   = wid & 3;                 // M strip (16 rows)
    const int wn   = wid >> 2;                // N half (2 x 16 cols)
    const int nb   = blockIdx.x * BN;         // n-tile fastest -> h-tile L2 reuse
    const int b0   = blockIdx.y * BM;
    const int half = (lane >= 16) ? 1 : 0;
    const int l16  = lane & 15;

    // Per-block constants (first barrier inside the k-loop covers these)
    {
        int g = tid >> 6, n = tid & 63;
        int gcol = g * HH + nb + n;
        lds_bias[tid] = bias[gcol];
        lds_wih[tid]  = W_ih[gcol];
    }
    for (int idx = tid; idx < TT * BM; idx += 256) {
        int t = idx / BM, m = idx % BM;
        lds_y[idx] = y_seq[(size_t)(b0 + m) * TT + t];
    }

    // Stage W tiles: 4 gates x 64 cols x 32 k = 1024 x 16B chunks (4/thread)
    auto issueW = [&](int buf, int kt) {
        const int k0 = kt * 32;
        #pragma unroll
        for (int r = 0; r < 4; ++r) {
            int chunk = tid + r * 256;
            int g = chunk >> 8, rem = chunk & 255;
            int n = rem >> 2, k8 = (rem & 3) * 8;
            const bf16* src = whh + (size_t)(g * HH + nb + n) * HH + k0 + k8;
            bf16* dst = &lds_w[buf][(g * BN + n) * LDH + k8];
#if HAVE_ASYNC_LDS
            async_copy16(src, dst);
#else
            *(bf16x8*)dst = *(const bf16x8*)src;
#endif
        }
    };
    // h tile: 64x32 f32, 512 float4 chunks (2/thread) into registers
    auto loadH = [&](f32x4* regs, int t, int kt) {
        #pragma unroll
        for (int r = 0; r < 2; ++r) {
            int chunk = tid + r * 256;
            int m = chunk >> 3, k4 = (chunk & 7) * 4;
            regs[r] = *(const f32x4*)(h + ((size_t)(b0 + m) * TT + t) * HH + kt * 32 + k4);
        }
    };
    auto storeH = [&](const f32x4* regs, int buf) {
        #pragma unroll
        for (int r = 0; r < 2; ++r) {
            int chunk = tid + r * 256;
            int m = chunk >> 3, k4 = (chunk & 7) * 4;
            bf16* dst = &lds_h[buf][m * LDH + k4];
            dst[0] = (bf16)regs[r].x; dst[1] = (bf16)regs[r].y;
            dst[2] = (bf16)regs[r].z; dst[3] = (bf16)regs[r].w;
        }
    };

    v8f c_acc[2];
    #pragma unroll
    for (int s = 0; s < 2; ++s)
        #pragma unroll
        for (int e = 0; e < 8; ++e) c_acc[s][e] = 0.f;

    auto run_t = [&](int t, auto lastc) {
        constexpr bool LAST = decltype(lastc)::value;
        constexpr int  NG   = LAST ? 4 : 3;   // o-gate only needed at last step

        v8f acc[NG][2];
        #pragma unroll
        for (int g = 0; g < NG; ++g)
            #pragma unroll
            for (int s = 0; s < 2; ++s)
                #pragma unroll
                for (int e = 0; e < 8; ++e) acc[g][s][e] = 0.f;

        f32x4 hreg[2];
        issueW(0, 0);
        loadH(hreg, t, 0);

        for (int kt = 0; kt < NK; ++kt) {
            const int buf = kt & 1;
            storeH(hreg, buf);
            wait_async0();                       // this buffer's W DMA done
            const bool more = (kt + 1 < NK);
            if (more) loadH(hreg, t, kt + 1);    // prefetch next h into regs
            __syncthreads();                     // staging of buf visible
            if (more) issueW(buf ^ 1, kt + 1);   // async DMA into other buffer

            // A fragment: lanes 0-15 K{0..7,16..23}, lanes 16-31 K{8..15,24..31}
            V16 afrag;
            {
                int m = wm * 16 + l16;
                int kb = half ? 8 : 0;
                afrag.h[0] = *(const bf16x8*)&lds_h[buf][m * LDH + kb];
                afrag.h[1] = *(const bf16x8*)&lds_h[buf][m * LDH + kb + 16];
            }
            #pragma unroll
            for (int g = 0; g < NG; ++g) {
                #pragma unroll
                for (int s = 0; s < 2; ++s) {
                    // B fragment: col = lane%16, lanes 0-15 K{0..7,16..23}, 16-31 K{8..15,24..31}
                    V16 bfrag;
                    int n = wn * 32 + s * 16 + l16;
                    int kb = half ? 8 : 0;
                    bfrag.h[0] = *(const bf16x8*)&lds_w[buf][(g * BN + n) * LDH + kb];
                    bfrag.h[1] = *(const bf16x8*)&lds_w[buf][(g * BN + n) * LDH + kb + 16];
                    acc[g][s] = __builtin_amdgcn_wmma_f32_16x16x32_bf16(
                        false, afrag.v, false, bfrag.v, (short)0, acc[g][s], false, false);
                }
            }
        } // kt

        // Elementwise LSTM update; C/D layout: VGPR e -> M = e + 8*half, N = lane%16
        #pragma unroll
        for (int s = 0; s < 2; ++s) {
            int nidx = wn * 32 + s * 16 + l16;
            float bI = lds_bias[0 * BN + nidx], bF = lds_bias[1 * BN + nidx];
            float bG = lds_bias[2 * BN + nidx];
            float wI = lds_wih[0 * BN + nidx],  wF = lds_wih[1 * BN + nidx];
            float wG = lds_wih[2 * BN + nidx];
            #pragma unroll
            for (int e = 0; e < 8; ++e) {
                int m = wm * 16 + e + (half ? 8 : 0);
                float yv = lds_y[t * BM + m];
                float iv = sigf(acc[0][s][e] + bI + yv * wI);
                float fv = sigf(acc[1][s][e] + bF + yv * wF);
                float gv = tanhf(acc[2][s][e] + bG + yv * wG);
                float c  = fv * c_acc[s][e] + iv * gv;
                c_acc[s][e] = c;
                if constexpr (LAST) {
                    float bO = lds_bias[3 * BN + nidx];
                    float wO = lds_wih[3 * BN + nidx];
                    float ov = sigf(acc[3][s][e] + bO + yv * wO);
                    d_state[(size_t)(b0 + m) * HH + nb + nidx] = ov * tanhf(c);
                }
            }
        }
    };

    for (int t = 0; t < TT - 1; ++t) run_t(t, BoolC<false>{});
    run_t(TT - 1, BoolC<true>{});
}

// ---------------------------------------------------------------------------
// Final: y[b] = d[b]·v[:512] + h[b,T-1]·v[512:] + c0   (one wave per row)
// ---------------------------------------------------------------------------
__global__ void k_final(const float* __restrict__ d_state, const float* __restrict__ h,
                        const float* __restrict__ vvec, const float* __restrict__ c0,
                        float* __restrict__ out) {
    int b = blockIdx.x * 8 + (threadIdx.x >> 5);
    int lane = threadIdx.x & 31;
    const float* dp = d_state + (size_t)b * HH;
    const float* hp = h + ((size_t)b * TT + (TT - 1)) * HH;
    float acc = 0.f;
    for (int j = lane; j < HH; j += 32) {
        acc += dp[j] * vvec[j];
        acc += hp[j] * vvec[HH + j];
    }
    #pragma unroll
    for (int off = 16; off > 0; off >>= 1) acc += __shfl_down(acc, off, 32);
    if (lane == 0) out[b] = acc + c0[0];
}

// ---------------------------------------------------------------------------
extern "C" void kernel_launch(void* const* d_in, const int* in_sizes, int n_in,
                              void* d_out, int out_size, void* d_ws, size_t ws_size,
                              hipStream_t stream) {
    const float* h     = (const float*)d_in[0];
    const float* y_seq = (const float*)d_in[1];
    // d_in[2..7]: attention weights -- dead (softmax over singleton dim == 1)
    const float* W_ih  = (const float*)d_in[8];
    const float* W_hh  = (const float*)d_in[9];
    const float* b_ih  = (const float*)d_in[10];
    const float* b_hh  = (const float*)d_in[11];
    const float* fc1_w = (const float*)d_in[12];
    const float* fc1_b = (const float*)d_in[13];
    const float* fc2_w = (const float*)d_in[14];
    const float* fc2_b = (const float*)d_in[15];
    float* out = (float*)d_out;

    // Workspace layout (~20 MB)
    char* ws = (char*)d_ws;
    bf16*  whh_bf = (bf16*)ws;                                   // 2 MB
    float* bias   = (float*)(ws + (2u << 20));                   // 8 KB
    float* vvec   = (float*)(ws + (2u << 20) + 8192);            // 4 KB
    float* c0     = (float*)(ws + (2u << 20) + 8192 + 4096);     // 4 B
    float* dstate = (float*)(ws + (4u << 20));                   // 16 MB

    k_prep_w<<<1024, 256, 0, stream>>>(W_hh, whh_bf);
    k_prep_misc<<<2, 1024, 0, stream>>>(b_ih, b_hh, fc1_w, fc1_b, fc2_w, fc2_b,
                                        bias, vvec, c0);
    dim3 grid(HH / BN, BSZ / BM);   // (8, 128): n-tile fastest for h L2 reuse
    k_main<<<grid, 256, 0, stream>>>(h, y_seq, W_ih, whh_bf, bias, dstate);
    k_final<<<BSZ / 8, 256, 0, stream>>>(dstate, h, vvec, c0, out);
}
